// DensityModulatedAttention_85478439125168
// MI455X (gfx1250) — matrix operand order repaired and checked by reading, not compiled, not run on hardware
//
#include <hip/hip_runtime.h>
#include <hip/hip_bf16.h>
#include <cstdint>

// Problem constants (match reference)
#define B_  2
#define L_  2048
#define D_  768
#define H_  12
#define DH_ 64

typedef __bf16 bf16;
typedef __attribute__((ext_vector_type(16))) __bf16 v16bf;
typedef __attribute__((ext_vector_type(8)))  __bf16 v8bf;
typedef __attribute__((ext_vector_type(8)))  float  v8f;
typedef __attribute__((ext_vector_type(4)))  unsigned int v4u;
typedef __attribute__((ext_vector_type(8)))  int    v8i;
typedef __attribute__((ext_vector_type(4)))  int    v4i;

// ---------------------------------------------------------------------------
// WMMA fragment helpers (wave32; A/B use this file's contiguous k order, C/D per CDNA5 ISA 7.12.2)
// A (16x32 bf16): lane holds row M = lane&15; elems e -> K = 16*(lane>=16)+e.
// B (32x16 bf16): lane holds col N = lane&15; elems e -> K = 16*(lane>=16)+e.
// C/D (16x16 f32): lane holds col N = lane&15; vgpr r -> M = r + 8*(lane>=16).
// ---------------------------------------------------------------------------
__device__ __forceinline__ v16bf frag_a16(const bf16* base, int lda, int lane) {
  const bf16* p = base + (lane & 15) * lda + ((lane >> 4) << 4);
  v8bf lo = *(const v8bf*)p;
  v8bf hi = *(const v8bf*)(p + 8);
  v16bf a;
#pragma unroll
  for (int e = 0; e < 8; ++e) { a[e] = lo[e]; a[8 + e] = hi[e]; }
  return a;
}

__device__ __forceinline__ v16bf frag_b16(const bf16* base, int ldb, int lane) {
  const bf16* p = base + (lane & 15) * ldb + ((lane >> 4) << 4);
  v8bf lo = *(const v8bf*)p;
  v8bf hi = *(const v8bf*)(p + 8);
  v16bf b;
#pragma unroll
  for (int e = 0; e < 8; ++e) { b[e] = lo[e]; b[8 + e] = hi[e]; }
  return b;
}

__device__ __forceinline__ v8f wmma_bf16(v16bf a, v16bf b, v8f c) {
  return __builtin_amdgcn_wmma_f32_16x16x32_bf16(false, a, false, b, (short)0, c,
                                                 false, false);
}

// ---------------------------------------------------------------------------
// 16-lane reductions in pure VALU via DPP (avoids ds_bpermute that __shfl_xor
// lowers to). xor1 = quad_perm(1,0,3,2)=0xB1; xor2 = quad_perm(2,3,0,1)=0x4E;
// after those, quads are uniform, so row_half_mirror(0x141) acts as xor4 and
// row_mirror(0x140) as xor8. Stays inside each 16-lane half of the wave,
// which is exactly the C-layout row group. DPP ctrl must be an ICE -> template.
// ---------------------------------------------------------------------------
template <int CTRL>
__device__ __forceinline__ float dpp_bcast(float x) {
  return __int_as_float(
      __builtin_amdgcn_mov_dpp(__float_as_int(x), CTRL, 0xF, 0xF, true));
}
__device__ __forceinline__ float red_max16(float v) {
  v = fmaxf(v, dpp_bcast<0xB1>(v));   // quad_perm(1,0,3,2)  == xor1
  v = fmaxf(v, dpp_bcast<0x4E>(v));   // quad_perm(2,3,0,1)  == xor2
  v = fmaxf(v, dpp_bcast<0x141>(v));  // row_half_mirror     (acts as xor4)
  v = fmaxf(v, dpp_bcast<0x140>(v));  // row_mirror          (acts as xor8)
  return v;
}
__device__ __forceinline__ float red_sum16(float v) {
  v += dpp_bcast<0xB1>(v);
  v += dpp_bcast<0x4E>(v);
  v += dpp_bcast<0x141>(v);
  v += dpp_bcast<0x140>(v);
  return v;
}

// ---------------------------------------------------------------------------
// Tensor Data Mover: 64x64-element (bf16) 2D tile -> LDS, with 4-DWORD padding
// after every 32 DWORDs (= one 64-elem row) => LDS row pitch 72 elems (144 B),
// which is also bank-conflict-free for the fragment reads (gcd(36,64)=4).
// D# fields per CDNA5 ISA.
// ---------------------------------------------------------------------------
__device__ __forceinline__ void tdm_load_tile64(uint32_t lds_addr,
                                                const void* gaddr,
                                                uint32_t tensor_d0,
                                                uint32_t tensor_d1,
                                                uint64_t stride0) {
  uint64_t ga = (uint64_t)(uintptr_t)gaddr;
  v4u g0;
  g0[0] = 1u;                                   // count=1, user descriptor
  g0[1] = lds_addr;                             // lds_addr [63:32]
  g0[2] = (uint32_t)ga;                         // global_addr low
  g0[3] = (uint32_t)(ga >> 32) | (2u << 30);    // global_addr hi | type=2
  v8i g1;
  g1[0] = (int)((1u << 16)      // data_size = 2 bytes
              | (1u << 20)      // pad_enable
              | (4u << 22)      // pad_interval: every 32 DWORDs
              | (3u << 25));    // pad_amount: 4 DWORDs
  g1[1] = (int)(tensor_d0 << 16);                        // tensor_dim0 [79:48]
  g1[2] = (int)((tensor_d0 >> 16) | (tensor_d1 << 16));  // tensor_dim1 [111:80]
  g1[3] = (int)((tensor_d1 >> 16) | (64u << 16));        // tile_dim0 = 64
  g1[4] = (int)64;                                       // tile_dim1=64, dim2=0
  g1[5] = (int)(uint32_t)stride0;                        // dim0 stride lo
  g1[6] = (int)((uint32_t)(stride0 >> 32) & 0xFFFFu);    // dim0 stride hi
  g1[7] = 0;
  v4i z = {0, 0, 0, 0};
#if __clang_major__ >= 23
  v8i z8 = {0, 0, 0, 0, 0, 0, 0, 0};
  __builtin_amdgcn_tensor_load_to_lds(g0, g1, z, z, z8, 0);
#else
  __builtin_amdgcn_tensor_load_to_lds(g0, g1, z, z, 0);
#endif
}

// ---------------------------------------------------------------------------
// Tiled GEMM:  C[M,N] = A[M,K] @ W[N,K]^T (+ bias[n])
// 256 threads = 8 waves; block tile 128x64, K step 32; each wave 32x32.
// LDS tiles filled with vectorized v8bf stores (packed f32->bf16 converts).
// ---------------------------------------------------------------------------
template <bool A_BF16>
__global__ __launch_bounds__(256) void gemm_bf16_wmma(
    const void* __restrict__ Aptr, const float* __restrict__ W,
    float* __restrict__ C, const float* __restrict__ bias, int M, int N, int K) {
  constexpr int BM = 128, BN = 64, BK = 32, LDA = BK + 8;  // 80B pitch, 16B ok
  __shared__ bf16 As[BM * LDA];
  __shared__ bf16 Bs[BN * LDA];
  const int tid  = threadIdx.x;
  const int lane = tid & 31;
  const int wave = tid >> 5;
  const int bm = blockIdx.x * BM;
  const int bn = blockIdx.y * BN;
  const int wm = (wave >> 1) * 32;   // 0..96
  const int wn = (wave & 1) * 32;    // 0..32

  v8f acc[2][2] = {};

  for (int k0 = 0; k0 < K; k0 += BK) {
    __syncthreads();
    {  // A tile: 128x32, 16 elems/thread
      int r = tid >> 1, c0 = (tid & 1) * 16;
      bf16* dst = As + r * LDA + c0;
      if (A_BF16) {
        const bf16* src = (const bf16*)Aptr + (size_t)(bm + r) * K + k0 + c0;
        *(v8bf*)dst = *(const v8bf*)src;
        *(v8bf*)(dst + 8) = *(const v8bf*)(src + 8);
      } else {
        const float* src = (const float*)Aptr + (size_t)(bm + r) * K + k0 + c0;
        v8bf t0, t1;
#pragma unroll
        for (int e = 0; e < 8; ++e) { t0[e] = (bf16)src[e]; t1[e] = (bf16)src[8 + e]; }
        *(v8bf*)dst = t0;
        *(v8bf*)(dst + 8) = t1;
      }
    }
    if (tid < 128) {  // B tile: 64x32
      int r = tid >> 1, c0 = (tid & 1) * 16;
      const float* src = W + (size_t)(bn + r) * K + k0 + c0;
      v8bf t0, t1;
#pragma unroll
      for (int e = 0; e < 8; ++e) { t0[e] = (bf16)src[e]; t1[e] = (bf16)src[8 + e]; }
      bf16* dst = Bs + r * LDA + c0;
      *(v8bf*)dst = t0;
      *(v8bf*)(dst + 8) = t1;
    }
    __syncthreads();
    v16bf af[2], bfr[2];
#pragma unroll
    for (int i = 0; i < 2; ++i) af[i] = frag_a16(As + (wm + 16 * i) * LDA, LDA, lane);
#pragma unroll
    for (int j = 0; j < 2; ++j) bfr[j] = frag_b16(Bs + (wn + 16 * j) * LDA, LDA, lane);
#pragma unroll
    for (int i = 0; i < 2; ++i)
#pragma unroll
      for (int j = 0; j < 2; ++j) acc[i][j] = wmma_bf16(af[i], bfr[j], acc[i][j]);
  }

  const int rn = lane & 15;
  const int rg = lane >> 4;
#pragma unroll
  for (int i = 0; i < 2; ++i)
#pragma unroll
    for (int j = 0; j < 2; ++j) {
      int n = bn + wn + 16 * j + rn;
      float bv = bias ? bias[n] : 0.f;
#pragma unroll
      for (int r = 0; r < 8; ++r) {
        int m = bm + wm + 16 * i + r + 8 * rg;
        C[(size_t)m * N + n] = acc[i][j][r] + bv;
      }
    }
}

// ---------------------------------------------------------------------------
// RMSNorm(q,k) + RoPE + layout change. One wave per (b,h,l) row, lane = pair.
// Q gets the attention scale (1/sqrt(64)) folded in here, so the flash kernel
// does no per-score scaling. V is written transposed (B,H,64,L).
// ---------------------------------------------------------------------------
__global__ __launch_bounds__(128) void qkv_norm_rope(
    const float* __restrict__ qkv, const float* __restrict__ pe,
    const float* __restrict__ qs, const float* __restrict__ ks,
    bf16* __restrict__ Q, bf16* __restrict__ Kn, bf16* __restrict__ Vt) {
  int gw   = (blockIdx.x * 128 + threadIdx.x) >> 5;  // (b,h,l) index
  int lane = threadIdx.x & 31;
  int l = gw % L_;
  int h = (gw / L_) % H_;
  int b = gw / (L_ * H_);
  const float* row = qkv + (size_t)(b * L_ + l) * (3 * D_) + h * DH_;
  int d0 = 2 * lane, d1 = d0 + 1;
  float q0 = row[d0],          q1 = row[d1];
  float k0 = row[D_ + d0],     k1 = row[D_ + d1];
  float v0 = row[2 * D_ + d0], v1 = row[2 * D_ + d1];
  float sq = q0 * q0 + q1 * q1;
  float sk = k0 * k0 + k1 * k1;
#pragma unroll
  for (int off = 16; off; off >>= 1) {  // wave32 reduction over head dim
    sq += __shfl_xor(sq, off, 32);
    sk += __shfl_xor(sk, off, 32);
  }
  float rq = rsqrtf(sq * (1.f / DH_) + 1e-6f);
  float rk = rsqrtf(sk * (1.f / DH_) + 1e-6f);
  q0 *= rq * qs[d0]; q1 *= rq * qs[d1];
  k0 *= rk * ks[d0]; k1 *= rk * ks[d1];
  const float* p = pe + ((size_t)l * (DH_ / 2) + lane) * 4;  // [l][pair][2][2]
  const float sc = 0.125f;  // 64^-0.5 folded into Q
  float qo0 = (p[0] * q0 + p[1] * q1) * sc;
  float qo1 = (p[2] * q0 + p[3] * q1) * sc;
  float ko0 = p[0] * k0 + p[1] * k1;
  float ko1 = p[2] * k0 + p[3] * k1;
  size_t base = ((size_t)(b * H_ + h) * L_ + l) * DH_;
  Q[base + d0]  = (bf16)qo0; Q[base + d1]  = (bf16)qo1;
  Kn[base + d0] = (bf16)ko0; Kn[base + d1] = (bf16)ko1;
  size_t vb = (size_t)(b * H_ + h) * DH_;
  Vt[(vb + d0) * L_ + l] = (bf16)v0;
  Vt[(vb + d1) * L_ + l] = (bf16)v1;
}

// ---------------------------------------------------------------------------
// Flash attention, WMMA bf16, fp32 accumulate. 4 waves/block, 16 queries/wave,
// 64-key chunks. K/V chunks are staged into LDS once per block by the Tensor
// Data Mover (double-buffered; issuing wave does s_wait_tensorcnt 0, then a
// block barrier publishes the buffer; next chunk issued right after so no
// reader still holds the target buffer). Row softmax stats use DPP-only
// 16-lane reductions. P is bounced C->A layout through wave-private LDS.
// NOTE: the reference's density bias is a per-query additive constant over all
// keys -> softmax-invariant; it cancels exactly and is intentionally dropped.
// ---------------------------------------------------------------------------
__global__ __launch_bounds__(128) void flash_attn_wmma(
    const bf16* __restrict__ Q, const bf16* __restrict__ Kn,
    const bf16* __restrict__ Vt, bf16* __restrict__ O) {
  constexpr int LDK = 72;  // 64 + 8 pad elems == TDM pad (4 DWORDs per row)
  __shared__ bf16 Ks[2][64 * LDK];
  __shared__ bf16 Vs[2][64 * LDK];
  __shared__ bf16 Plds[4][16 * LDK];
  const int lane  = threadIdx.x & 31;
  const int wave  = threadIdx.x >> 5;
  const int bh    = blockIdx.x / (L_ / 64);
  const int qbase = (blockIdx.x % (L_ / 64)) * 64 + wave * 16;
  const bf16* q  = Q  + ((size_t)bh * L_ + qbase) * DH_;
  const bf16* Kp = Kn + (size_t)bh * L_ * DH_;
  const bf16* Vp = Vt + (size_t)bh * DH_ * L_;

  // Q A-fragments for K-dim sections [0,32) and [32,64)
  v16bf aq[2];
  {
    const bf16* p = q + (lane & 15) * DH_ + ((lane >> 4) << 4);
#pragma unroll
    for (int t = 0; t < 2; ++t) {
      v8bf lo = *(const v8bf*)(p + 32 * t);
      v8bf hi = *(const v8bf*)(p + 32 * t + 8);
#pragma unroll
      for (int e = 0; e < 8; ++e) { aq[t][e] = lo[e]; aq[t][8 + e] = hi[e]; }
    }
  }

  v8f o[4] = {};
  float mrow[8], lrow[8];
#pragma unroll
  for (int r = 0; r < 8; ++r) { mrow[r] = -3.0e38f; lrow[r] = 0.f; }

  bf16* P = Plds[wave];

  if (wave == 0) {  // preload chunk 0
    tdm_load_tile64((uint32_t)(uintptr_t)&Ks[0][0], Kp, DH_, L_, DH_);
    tdm_load_tile64((uint32_t)(uintptr_t)&Vs[0][0], Vp, L_, DH_, L_);
  }

  for (int cb = 0; cb < L_ / 64; ++cb) {
    const int cur = cb & 1;
    if (wave == 0) __builtin_amdgcn_s_wait_tensorcnt(0);
    __syncthreads();  // chunk cb resident; everyone done reading buffer cur^1
    if (wave == 0 && cb + 1 < L_ / 64) {
      const int kb2 = (cb + 1) * 64;
      tdm_load_tile64((uint32_t)(uintptr_t)&Ks[cur ^ 1][0],
                      Kp + (size_t)kb2 * DH_, DH_, L_, DH_);
      tdm_load_tile64((uint32_t)(uintptr_t)&Vs[cur ^ 1][0], Vp + kb2, L_, DH_, L_);
    }
    const bf16* Kc = Ks[cur];
    const bf16* Vc = Vs[cur];

    // ---- S = q @ k^T for 16x64 key chunk (4 C-frags, 2 K-steps each) ----
    v8f s[4] = {};
#pragma unroll
    for (int t = 0; t < 4; ++t) {
#pragma unroll
      for (int ksec = 0; ksec < 2; ++ksec) {
        const bf16* kp =
            Kc + (16 * t + (lane & 15)) * LDK + ksec * 32 + ((lane >> 4) << 4);
        v8bf lo = *(const v8bf*)kp;
        v8bf hi = *(const v8bf*)(kp + 8);
        v16bf bb;
#pragma unroll
        for (int e = 0; e < 8; ++e) { bb[e] = lo[e]; bb[8 + e] = hi[e]; }
        s[t] = wmma_bf16(aq[ksec], bb, s[t]);
      }
    }
    // ---- online softmax over 64 keys ----
#pragma unroll
    for (int r = 0; r < 8; ++r) {
      float lm = fmaxf(fmaxf(s[0][r], s[1][r]), fmaxf(s[2][r], s[3][r]));
      float v  = red_max16(lm);
      float mn = fmaxf(mrow[r], v);
      float alpha = __expf(mrow[r] - mn);
      mrow[r] = mn;
      float p0 = __expf(s[0][r] - mn);
      float p1 = __expf(s[1][r] - mn);
      float p2 = __expf(s[2][r] - mn);
      float p3 = __expf(s[3][r] - mn);
      float ps = red_sum16((p0 + p1) + (p2 + p3));
      lrow[r] = lrow[r] * alpha + ps;
#pragma unroll
      for (int dt = 0; dt < 4; ++dt) o[dt][r] *= alpha;
      int prow = (r + 8 * (lane >> 4)) * LDK + (lane & 15);  // C -> row-major
      P[prow]      = (bf16)p0;
      P[prow + 16] = (bf16)p1;
      P[prow + 32] = (bf16)p2;
      P[prow + 48] = (bf16)p3;
    }
    asm volatile("s_wait_dscnt 0" ::: "memory");  // order P stores vs loads
    // ---- O += P @ V (two K=32 steps over the 64 keys) ----
    v16bf ap0 = frag_a16(P, LDK, lane);
    v16bf ap1 = frag_a16(P + 32, LDK, lane);
#pragma unroll
    for (int dt = 0; dt < 4; ++dt) {
#pragma unroll
      for (int ksec = 0; ksec < 2; ++ksec) {
        const bf16* vp =
            Vc + (dt * 16 + (lane & 15)) * LDK + ksec * 32 + ((lane >> 4) << 4);
        v8bf lo = *(const v8bf*)vp;
        v8bf hi = *(const v8bf*)(vp + 8);
        v16bf bv;
#pragma unroll
        for (int e = 0; e < 8; ++e) { bv[e] = lo[e]; bv[8 + e] = hi[e]; }
        o[dt] = wmma_bf16(ksec ? ap1 : ap0, bv, o[dt]);
      }
    }
  }

  // ---- epilogue: O/l -> attn buffer (B,L,D) bf16 ----
  const int b = bh / H_, h = bh % H_;
#pragma unroll
  for (int dt = 0; dt < 4; ++dt)
#pragma unroll
    for (int r = 0; r < 8; ++r) {
      int m = qbase + r + 8 * (lane >> 4);
      float val = o[dt][r] / lrow[r];
      O[(size_t)(b * L_ + m) * D_ + h * DH_ + dt * 16 + (lane & 15)] = (bf16)val;
    }
}

// ---------------------------------------------------------------------------
// Launch
// ---------------------------------------------------------------------------
extern "C" void kernel_launch(void* const* d_in, const int* in_sizes, int n_in,
                              void* d_out, int out_size, void* d_ws, size_t ws_size,
                              hipStream_t stream) {
  (void)in_sizes; (void)n_in; (void)out_size; (void)ws_size;
  const float* x      = (const float*)d_in[0];
  // d_in[1] density_weights, d_in[8] density_scale, d_in[9] density_bias:
  // per-query additive score bias -> cancels exactly in softmax; unused.
  const float* pe     = (const float*)d_in[2];
  const float* qkv_w  = (const float*)d_in[3];
  const float* q_s    = (const float*)d_in[4];
  const float* k_s    = (const float*)d_in[5];
  const float* proj_w = (const float*)d_in[6];
  const float* proj_b = (const float*)d_in[7];
  float* out = (float*)d_out;

  char* ws = (char*)d_ws;
  float* qkv = (float*)ws;                       // 4096*2304 f32  = 37,748,736 B
  bf16* qn   = (bf16*)(ws + 37748736);           // B*H*L*64 bf16  =  6,291,456 B
  bf16* kn   = (bf16*)(ws + 44040192);           //                   6,291,456 B
  bf16* vT   = (bf16*)(ws + 50331648);           // transposed V      6,291,456 B
  bf16* attn = (bf16*)(ws + 56623104);           // B*L*D bf16        6,291,456 B

  // 1) qkv = x @ qkv_w^T     (M=4096, N=2304, K=768)
  {
    dim3 g((B_ * L_) / 128, (3 * D_) / 64);
    gemm_bf16_wmma<false><<<g, 256, 0, stream>>>(x, qkv_w, qkv, nullptr,
                                                 B_ * L_, 3 * D_, D_);
  }
  // 2) RMSNorm + RoPE + relayout (one wave per (b,h,l))
  {
    int waves = B_ * H_ * L_;
    qkv_norm_rope<<<waves / 4, 128, 0, stream>>>(qkv, pe, q_s, k_s, qn, kn, vT);
  }
  // 3) flash attention (64 queries / block, TDM-staged K/V)
  {
    flash_attn_wmma<<<B_ * H_ * (L_ / 64), 128, 0, stream>>>(qn, kn, vT, attn);
  }
  // 4) out = attn @ proj_w^T + proj_b   (M=4096, N=768, K=768)
  {
    dim3 g((B_ * L_) / 128, D_ / 64);
    gemm_bf16_wmma<true><<<g, 256, 0, stream>>>(attn, proj_w, out, proj_b,
                                                B_ * L_, D_, D_);
  }
}
